// TransitionUp_67439576482095
// MI455X (gfx1250) — matrix operand order repaired and checked by reading, not compiled, not run on hardware
//
#include <hip/hip_runtime.h>
#include <hip/hip_bf16.h>

// ---------------------------------------------------------------------------
// Problem constants (match reference)
// ---------------------------------------------------------------------------
constexpr int CH    = 256;      // channels
constexpr int NPTS  = 131072;   // points
constexpr int NSEG  = 16;       // segments
constexpr float EPS = 1e-5f;

// GEMM tiling
constexpr int TM   = 128;       // rows per block (8 waves x 16)
constexpr int TN   = 64;        // cols per block (4 wmma tiles per wave)
constexpr int BSTR = 264;       // LDS stride in halfs: 528B row -> 4-bank rotate/row

// workspace layout (in floats)
constexpr int WS_SUMS  = 0;                 // [16,256] segment sums  (zeroed)
constexpr int WS_GSUM  = WS_SUMS + NSEG*CH; // [256] column sums      (zeroed)
constexpr int WS_GSQ   = WS_GSUM + CH;      // [256] column sumsq     (zeroed)
constexpr int WS_Z     = WS_GSQ  + CH;      // [16,256] per-seg bias z
constexpr int WS_SCALE = WS_Z + NSEG*CH;    // [256]
constexpr int WS_SHIFT = WS_SCALE + CH;     // [256]

typedef __attribute__((ext_vector_type(16))) __bf16 v16bf;
typedef __attribute__((ext_vector_type(8)))  __bf16 v8bf;
typedef __attribute__((ext_vector_type(8)))  float  v8f;

// Scheduler fence: forbid the machine scheduler from sinking loads past this
// point (keeps the batched B fragments in distinct registers so WMMAs don't
// serialize on s_wait_dscnt 0).
#if defined(__has_builtin)
#  if __has_builtin(__builtin_amdgcn_sched_barrier)
#    define SCHED_FENCE() __builtin_amdgcn_sched_barrier(0)
#  endif
#endif
#ifndef SCHED_FENCE
#  define SCHED_FENCE()
#endif

// ---------------------------------------------------------------------------
// 1) Partial per-segment sums of x (rows sorted by segment): 512 blocks,
//    thread t owns channel t; fully coalesced 1KB-row loads; flush on
//    segment crossings with global f32 atomics.
// ---------------------------------------------------------------------------
__global__ __launch_bounds__(256) void seg_sum_kernel(
    const float* __restrict__ x, const int* __restrict__ o,
    float* __restrict__ sums) {
  const int t  = threadIdx.x;
  const int r0 = blockIdx.x * 256;
  int s = 0;
  while (s < NSEG - 1 && o[s] <= r0) s++;
  float acc = 0.f;
  for (int r = r0; r < r0 + 256; ++r) {
    if (r >= o[s]) {                      // crossed a segment boundary
      atomicAdd(&sums[s * CH + t], acc);
      acc = 0.f;
      while (s < NSEG - 1 && o[s] <= r) s++;
    }
    acc += x[(size_t)r * CH + t];
  }
  atomicAdd(&sums[s * CH + t], acc);
}

// ---------------------------------------------------------------------------
// 2) Tiny per-segment math: means -> y = relu(means@W2+b2) -> z = y@W1_bot+b1
//    One block, thread t owns output column t. ~4 MFLOP total.
// ---------------------------------------------------------------------------
__global__ __launch_bounds__(256) void small_gemm_kernel(
    const float* __restrict__ sums, const int* __restrict__ o,
    const float* __restrict__ W2, const float* __restrict__ b2,
    const float* __restrict__ W1, const float* __restrict__ b1,
    float* __restrict__ z) {
  __shared__ float means[NSEG][CH];
  __shared__ float ybuf[NSEG][CH];
  const int t = threadIdx.x;
  for (int s = 0; s < NSEG; ++s) {
    const int cnt = o[s] - (s ? o[s - 1] : 0);
    means[s][t] = sums[s * CH + t] / (float)cnt;
  }
  __syncthreads();
  for (int s = 0; s < NSEG; ++s) {
    float acc = b2[t];
    for (int k = 0; k < CH; ++k) acc = fmaf(means[s][k], W2[k * CH + t], acc);
    ybuf[s][t] = fmaxf(acc, 0.f);
  }
  __syncthreads();
  for (int s = 0; s < NSEG; ++s) {
    float acc = b1[t];
    for (int k = 0; k < CH; ++k) acc = fmaf(ybuf[s][k], W1[(size_t)(CH + k) * CH + t], acc);
    z[s * CH + t] = acc;
  }
}

// ---------------------------------------------------------------------------
// 3) Main fused kernel: h = x @ W1_top + z[seg]  via v_wmma_f32_16x16x32_bf16,
//    plus per-column sum / sumsq (LDS ds_add_f32 -> global_atomic_add_f32).
//    Block = 256 threads = 8 waves; wave computes 16 rows x 64 cols.
//    Software pipelined: per K-step, all 8 ds_load_b128 (4 B fragments) and
//    the next A fragment's global loads + bf16 converts are issued BEFORE a
//    sched_barrier, then 4 WMMAs run back-to-back behind partial waits.
// ---------------------------------------------------------------------------
struct AFrag { union { v16bf v; __bf16 e[16]; }; };
struct BFrag { union { v16bf v; v8bf half8[2]; }; };

__device__ __forceinline__ void load_a_frag(const float* __restrict__ xrow,
                                            int kk, int k0off, AFrag& a) {
  const float4 f0 = *(const float4*)(xrow + kk + k0off);
  const float4 f1 = *(const float4*)(xrow + kk + k0off + 4);
  const float4 f2 = *(const float4*)(xrow + kk + k0off + 16);
  const float4 f3 = *(const float4*)(xrow + kk + k0off + 20);
  a.e[0]=(__bf16)f0.x;  a.e[1]=(__bf16)f0.y;  a.e[2]=(__bf16)f0.z;  a.e[3]=(__bf16)f0.w;
  a.e[4]=(__bf16)f1.x;  a.e[5]=(__bf16)f1.y;  a.e[6]=(__bf16)f1.z;  a.e[7]=(__bf16)f1.w;
  a.e[8]=(__bf16)f2.x;  a.e[9]=(__bf16)f2.y;  a.e[10]=(__bf16)f2.z; a.e[11]=(__bf16)f2.w;
  a.e[12]=(__bf16)f3.x; a.e[13]=(__bf16)f3.y; a.e[14]=(__bf16)f3.z; a.e[15]=(__bf16)f3.w;
}

__global__ __launch_bounds__(256) void gemm_wmma_kernel(
    const float* __restrict__ x, const int* __restrict__ o,
    const float* __restrict__ W1, const float* __restrict__ z,
    float* __restrict__ h, float* __restrict__ gsum, float* __restrict__ gsq) {
  __shared__ __align__(16) __bf16 Bs[TN * BSTR];  // W1_top tile, [col][k], bf16
  __shared__ float colsum[TN];
  __shared__ float colsq[TN];

  const int tid    = threadIdx.x;
  const int lane   = tid & 31;          // wave32
  const int wave   = tid >> 5;          // 0..7
  const int blockN = blockIdx.x * TN;
  const int blockM = blockIdx.y * TM;

  // stage W1_top tile into LDS, transposed to [col][k] so B-fragment reads are
  // two 8-half runs at k+0 and k+16 (ds_load_b128 x2); padded stride kills bank conflicts.
  for (int idx = tid; idx < TN * CH; idx += 256) {
    const int c = idx & (TN - 1);
    const int k = idx >> 6;             // TN == 64
    Bs[c * BSTR + k] = (__bf16)W1[(size_t)k * CH + blockN + c];
  }
  if (tid < TN) { colsum[tid] = 0.f; colsq[tid] = 0.f; }
  __syncthreads();

  const int kh      = lane >> 4;        // K-half select (A/B fragment layouts)
  const int n       = lane & 15;        // column within 16-wide tile
  const int rowBase = blockM + wave * 16;
  const int mBase   = rowBase + 8 * kh; // C/D layout: lanes16-31 hold M=8..15

  // Accumulators initialised with per-segment bias z[seg(m)][col]
  v8f acc[4];
  {
    int s = 0;
    #pragma unroll
    for (int v = 0; v < 8; ++v) {
      const int mv = mBase + v;
      while (s < NSEG - 1 && o[s] <= mv) s++;
      #pragma unroll
      for (int t = 0; t < 4; ++t)
        acc[t][v] = z[s * CH + blockN + t * 16 + n];
    }
  }

  // A layout (16-bit 16x32): lane = row; lanes 0-15 hold K {0..7,16..23},
  // lanes 16-31 hold K {8..15,24..31}. Load fp32 from global, convert to bf16.
  const float* xrow  = x + (size_t)(rowBase + n) * CH;
  const int    k0off = 8 * kh;

  AFrag a_cur, a_nxt;
  load_a_frag(xrow, 0, k0off, a_cur);

  #pragma unroll
  for (int kk = 0; kk < CH; kk += 32) {
    // ---- batch-issue all 4 B fragments for this K-step
    BFrag b[4];
    #pragma unroll
    for (int t = 0; t < 4; ++t) {
      const __bf16* bp = &Bs[(t * 16 + n) * BSTR + kk + 8 * kh];
      b[t].half8[0] = *(const v8bf*)(bp);
      b[t].half8[1] = *(const v8bf*)(bp + 16);
    }
    // ---- prefetch + load/convert the NEXT A fragment into the alternate
    //      register set (hides VMEM latency; keeps converts off live WMMA srcs)
    if (kk + 32 < CH) {
      __builtin_prefetch(xrow + kk + 64 + k0off);
      load_a_frag(xrow, kk + 32, k0off, a_nxt);
    }
    // ---- fence: loads above may NOT sink below this point, so all four B
    //      fragments stay live in distinct registers and the WMMAs only pay
    //      partial dscnt waits instead of a full LDS round-trip each.
    SCHED_FENCE();
    #pragma unroll
    for (int t = 0; t < 4; ++t) {
      acc[t] = __builtin_amdgcn_wmma_f32_16x16x32_bf16(
          /*neg_a=*/false, a_cur.v, /*neg_b=*/false, b[t].v,
          /*c_mod=*/(short)0, acc[t], /*reuse_a=*/false, /*reuse_b=*/false);
    }
    a_cur = a_nxt;   // register rename; folded away by full unroll
  }

  // Epilogue: write h tile and fold per-column stats (sum, sumsq).
  #pragma unroll
  for (int t = 0; t < 4; ++t) {
    const int c = blockN + t * 16 + n;
    float ls = 0.f, lq = 0.f;
    #pragma unroll
    for (int v = 0; v < 8; ++v) {
      const float hv = acc[t][v];
      h[(size_t)(mBase + v) * CH + c] = hv;
      ls += hv;
      lq += hv * hv;
    }
    atomicAdd(&colsum[t * 16 + n], ls);   // ds_add_f32
    atomicAdd(&colsq[t * 16 + n], lq);
  }
  __syncthreads();
  if (tid < TN) {
    atomicAdd(&gsum[blockN + tid], colsum[tid]);  // global_atomic_add_f32
    atomicAdd(&gsq[blockN + tid],  colsq[tid]);
  }
}

// ---------------------------------------------------------------------------
// 4) Fold column stats into BN scale/shift.
// ---------------------------------------------------------------------------
__global__ __launch_bounds__(256) void bn_stats_kernel(
    const float* __restrict__ gsum, const float* __restrict__ gsq,
    const float* __restrict__ gamma, const float* __restrict__ beta,
    float* __restrict__ scale, float* __restrict__ shift) {
  const int t = threadIdx.x;
  const float inv_n = 1.f / (float)NPTS;
  const float mu  = gsum[t] * inv_n;
  const float var = gsq[t] * inv_n - mu * mu;
  const float sc  = gamma[t] * rsqrtf(var + EPS);
  scale[t] = sc;
  shift[t] = beta[t] - mu * sc;
}

// ---------------------------------------------------------------------------
// 5) In-place BN + ReLU over h (float4 per thread).
// ---------------------------------------------------------------------------
__global__ __launch_bounds__(256) void bn_apply_kernel(
    float* __restrict__ h, const float* __restrict__ scale,
    const float* __restrict__ shift) {
  const size_t idx = (size_t)blockIdx.x * blockDim.x + threadIdx.x; // float4 idx
  const int c0 = (int)(idx & (CH / 4 - 1)) * 4;
  float4 v = ((const float4*)h)[idx];
  v.x = fmaxf(fmaf(v.x, scale[c0 + 0], shift[c0 + 0]), 0.f);
  v.y = fmaxf(fmaf(v.y, scale[c0 + 1], shift[c0 + 1]), 0.f);
  v.z = fmaxf(fmaf(v.z, scale[c0 + 2], shift[c0 + 2]), 0.f);
  v.w = fmaxf(fmaf(v.w, scale[c0 + 3], shift[c0 + 3]), 0.f);
  ((float4*)h)[idx] = v;
}

// ---------------------------------------------------------------------------
extern "C" void kernel_launch(void* const* d_in, const int* in_sizes, int n_in,
                              void* d_out, int out_size, void* d_ws, size_t ws_size,
                              hipStream_t stream) {
  const float* x     = (const float*)d_in[0];
  const int*   o     = (const int*)  d_in[1];
  const float* W1    = (const float*)d_in[2];
  const float* b1    = (const float*)d_in[3];
  const float* W2    = (const float*)d_in[4];
  const float* b2    = (const float*)d_in[5];
  const float* gamma = (const float*)d_in[6];
  const float* beta  = (const float*)d_in[7];

  float* h  = (float*)d_out;          // [N, C] staged here, finished in place
  float* ws = (float*)d_ws;
  float* sums  = ws + WS_SUMS;
  float* gsum  = ws + WS_GSUM;
  float* gsq   = ws + WS_GSQ;
  float* z     = ws + WS_Z;
  float* scale = ws + WS_SCALE;
  float* shift = ws + WS_SHIFT;

  // zero atomic targets (sums, gsum, gsq are contiguous at ws start)
  hipMemsetAsync(ws, 0, (size_t)WS_Z * sizeof(float), stream);

  seg_sum_kernel<<<NPTS / 256, 256, 0, stream>>>(x, o, sums);
  small_gemm_kernel<<<1, 256, 0, stream>>>(sums, o, W2, b2, W1, b1, z);

  dim3 grid(CH / TN, NPTS / TM);      // (4, 1024)
  gemm_wmma_kernel<<<grid, 256, 0, stream>>>(x, o, W1, z, h, gsum, gsq);

  bn_stats_kernel<<<1, CH, 0, stream>>>(gsum, gsq, gamma, beta, scale, shift);
  bn_apply_kernel<<<(NPTS * CH / 4) / 256, 256, 0, stream>>>(h, scale, shift);
}
